// NeuralKYAttention_39865886441711
// MI455X (gfx1250) — matrix operand order repaired and checked by reading, not compiled, not run on hardware
//
#include <hip/hip_runtime.h>
#include <hip/hip_bf16.h>

typedef __attribute__((ext_vector_type(16))) _Float16 v16h;
typedef __attribute__((ext_vector_type(8)))  float    v8f;

// 32 bytes: 16 f16 == two float4 vector loads
typedef union { v16h v; float4 q[2]; } u16h;
typedef union { v8f  v; float  e[8]; } u8f;
typedef union { _Float16 h[4]; float2 f; } hpack4;

constexpr int kB   = 2;
constexpr int kL   = 2048;
constexpr int kD   = 1024;
constexpr int kH   = 16;
constexpr int kHD  = 64;
constexpr int kWIN = 15;
constexpr int kGH  = 128;
constexpr int kM   = kB * kL;            // 4096 GEMM rows
constexpr size_t kBLD = (size_t)kM * kD; // 4,194,304
constexpr size_t kDD  = (size_t)kD * kD; // 1,048,576

// ---------------------------------------------------------------------------
// Elementwise f32 -> f16 pre-conversion (one streaming pass; removes all
// cvt work and half the bytes from the GEMM hot loop).
// ---------------------------------------------------------------------------
__global__ __launch_bounds__(256) void f32_to_f16_kernel(
    const float* __restrict__ src, _Float16* __restrict__ dst, int n4)
{
  const int i = blockIdx.x * blockDim.x + threadIdx.x;
  if (i < n4) {
    const float4 f = ((const float4*)src)[i];
    hpack4 p;
    p.h[0] = (_Float16)f.x; p.h[1] = (_Float16)f.y;
    p.h[2] = (_Float16)f.z; p.h[3] = (_Float16)f.w;
    ((float2*)dst)[i] = p.f;
  }
}

// ---------------------------------------------------------------------------
// C(f32) = A(MxK,f16) * Bm(KxN,f16).  One wave computes a 32x64 tile:
// 2 A fragments x 4 B fragments -> 8 v_wmma_f32_16x16x32_f16 per k-step.
// Fragments are loaded directly in ISA register layout:
//   A 16x32 f16: lane half hf holds K = 8hf+{0..7} (dw0-3), 16+8hf+{0..7} (dw4-7)
//   B 32x16 f16: lane%16 = N col; halves hold K in the same order as A
// blockIdx.z selects among 3 (B,C) pairs so QKV runs as one launch.
// ---------------------------------------------------------------------------
__global__ __launch_bounds__(128) void gemm_f16wmma(
    const _Float16* __restrict__ A,
    const _Float16* __restrict__ B0, const _Float16* __restrict__ B1,
    const _Float16* __restrict__ B2,
    float* __restrict__ C0, float* __restrict__ C1, float* __restrict__ C2,
    int M, int N, int K)
{
  const _Float16* Bm = B0; float* Cm = C0;
  if (blockIdx.z == 1) { Bm = B1; Cm = C1; }
  if (blockIdx.z == 2) { Bm = B2; Cm = C2; }

  const int lane  = threadIdx.x & 31;
  const int wave  = blockIdx.x * (blockDim.x >> 5) + (threadIdx.x >> 5);
  const int ntile = N >> 6;                 // 64-wide column tiles
  const int tm    = wave / ntile;           // 32-row tile index
  const int tn    = wave % ntile;
  if (tm * 32 >= M) return;                 // wave-uniform guard

  const int r  = lane & 15;
  const int hf = lane >> 4;
  const _Float16* Arow0 = A + (size_t)(tm * 32 + r) * K;        // rows 0..15
  const _Float16* Arow1 = Arow0 + (size_t)16 * K;               // rows 16..31
  const int cb = tn * 64;

  v8f c00 = {}, c01 = {}, c02 = {}, c03 = {};
  v8f c10 = {}, c11 = {}, c12 = {}, c13 = {};

  for (int k0 = 0; k0 < K; k0 += 32) {
    u16h a0, a1;
    a0.q[0] = *(const float4*)(Arow0 + k0 + 8 * hf);
    a0.q[1] = *(const float4*)(Arow0 + k0 + 16 + 8 * hf);
    a1.q[0] = *(const float4*)(Arow1 + k0 + 8 * hf);
    a1.q[1] = *(const float4*)(Arow1 + k0 + 16 + 8 * hf);

    const _Float16* bk = Bm + (size_t)k0 * N + cb + r;   // lane%16 = N col
    u16h b0, b1, b2, b3;
    #pragma unroll
    for (int s = 0; s < 16; ++s) {
      const int kk = (s < 8) ? (8 * hf + s) : (16 + 8 * hf + (s - 8));
      const _Float16* p = bk + (size_t)kk * N;
      b0.v[s] = p[0]; b1.v[s] = p[16]; b2.v[s] = p[32]; b3.v[s] = p[48];
    }

    c00 = __builtin_amdgcn_wmma_f32_16x16x32_f16(false, a0.v, false, b0.v, (short)0, c00, false, false);
    c01 = __builtin_amdgcn_wmma_f32_16x16x32_f16(false, a0.v, false, b1.v, (short)0, c01, false, false);
    c02 = __builtin_amdgcn_wmma_f32_16x16x32_f16(false, a0.v, false, b2.v, (short)0, c02, false, false);
    c03 = __builtin_amdgcn_wmma_f32_16x16x32_f16(false, a0.v, false, b3.v, (short)0, c03, false, false);
    c10 = __builtin_amdgcn_wmma_f32_16x16x32_f16(false, a1.v, false, b0.v, (short)0, c10, false, false);
    c11 = __builtin_amdgcn_wmma_f32_16x16x32_f16(false, a1.v, false, b1.v, (short)0, c11, false, false);
    c12 = __builtin_amdgcn_wmma_f32_16x16x32_f16(false, a1.v, false, b2.v, (short)0, c12, false, false);
    c13 = __builtin_amdgcn_wmma_f32_16x16x32_f16(false, a1.v, false, b3.v, (short)0, c13, false, false);
  }

  // C/D layout: VGPR i -> M = i + 8*hf, col = r (+ subtile offset)
  u8f s0, s1, s2, s3;
  const int row0 = tm * 32 + 8 * hf;
  s0.v = c00; s1.v = c01; s2.v = c02; s3.v = c03;
  #pragma unroll
  for (int i = 0; i < 8; ++i) {
    float* crow = Cm + (size_t)(row0 + i) * N + (cb + r);
    crow[0] = s0.e[i]; crow[16] = s1.e[i]; crow[32] = s2.e[i]; crow[48] = s3.e[i];
  }
  s0.v = c10; s1.v = c11; s2.v = c12; s3.v = c13;
  #pragma unroll
  for (int i = 0; i < 8; ++i) {
    float* crow = Cm + (size_t)(row0 + 16 + i) * N + (cb + r);
    crow[0] = s0.e[i]; crow[16] = s1.e[i]; crow[32] = s2.e[i]; crow[48] = s3.e[i];
  }
}

// ---------------------------------------------------------------------------
// conn[h][w] = softmax_w( sum_j silu(pos_w*W1[h,j]) * (pos_w*W3[h,j]) * W2[h,j] )
// ---------------------------------------------------------------------------
__global__ __launch_bounds__(256) void gate_kernel(
    const float* __restrict__ W1, const float* __restrict__ W2,
    const float* __restrict__ W3, float* __restrict__ conn)
{
  __shared__ float raw[kH * kWIN];
  const int tid = threadIdx.x;
  if (tid < kH * kWIN) {
    const int h = tid / kWIN, w = tid % kWIN;
    const float pos = (float)w / (float)(kWIN - 1);
    float s = 0.f;
    for (int j = 0; j < kGH; ++j) {
      const float a = pos * W1[h * kGH + j];
      const float g = a / (1.f + expf(-a));          // silu
      s += g * (pos * W3[h * kGH + j]) * W2[h * kGH + j];
    }
    raw[tid] = s;
  }
  __syncthreads();
  if (tid < kH * kWIN) {
    const int h = tid / kWIN;
    float mx = -1e30f;
    for (int w = 0; w < kWIN; ++w) mx = fmaxf(mx, raw[h * kWIN + w]);
    float den = 0.f;
    for (int w = 0; w < kWIN; ++w) den += expf(raw[h * kWIN + w] - mx);
    conn[tid] = expf(raw[tid] - mx) / den;
  }
}

// ---------------------------------------------------------------------------
// One wave per (b,h,l); lane = rotary pair (HD/2 == 32 == wave32 lanes).
// Rotary fused inline for q and k. Invalid window slots get score 0 and zero
// v contribution (matches the reference's zeroed-kw/vw semantics exactly).
// Writes ctx directly as f16 in (b, l, h*HD+hd) layout -> A matrix of Wo GEMM.
// ---------------------------------------------------------------------------
__global__ __launch_bounds__(256) void attn_kernel(
    const float* __restrict__ q, const float* __restrict__ k, const float* __restrict__ v,
    const float* __restrict__ fcos, const float* __restrict__ fsin,
    const float* __restrict__ conn, _Float16* __restrict__ ctx)
{
  const int lane = threadIdx.x & 31;
  const int wid  = (int)((blockIdx.x * blockDim.x + threadIdx.x) >> 5);
  const int l  = wid % kL;
  const int bh = wid / kL;
  const int h  = bh % kH;
  const int b  = bh / kH;
  const size_t rowq = (size_t)(b * kL + l) * kD + h * kHD + 2 * lane;

  const float cq = fcos[l * (kHD / 2) + lane];
  const float sq = fsin[l * (kHD / 2) + lane];
  const float qr = q[rowq], qi = q[rowq + 1];
  const float Qr = qr * cq - qi * sq;
  const float Qi = qr * sq + qi * cq;

  float sc[kWIN];
  #pragma unroll
  for (int w = 0; w < kWIN; ++w) {
    const int l2 = l + w - (kWIN - 1);        // wave-uniform
    float t = 0.f;
    if (l2 >= 0) {
      const size_t rk = (size_t)(b * kL + l2) * kD + h * kHD + 2 * lane;
      const float kr = k[rk], ki = k[rk + 1];
      const float c2 = fcos[l2 * (kHD / 2) + lane];
      const float s2 = fsin[l2 * (kHD / 2) + lane];
      t = Qr * (kr * c2 - ki * s2) + Qi * (kr * s2 + ki * c2);
    }
    #pragma unroll
    for (int m = 16; m >= 1; m >>= 1) t += __shfl_xor(t, m, 32);
    sc[w] = (l2 >= 0) ? t * 0.125f : 0.0f;    // scale = 1/sqrt(64); zero, not -inf
  }

  float mx = sc[0];
  #pragma unroll
  for (int w = 1; w < kWIN; ++w) mx = fmaxf(mx, sc[w]);
  float ex[kWIN]; float den = 0.f;
  #pragma unroll
  for (int w = 0; w < kWIN; ++w) { ex[w] = expf(sc[w] - mx); den += ex[w]; }

  float fw[kWIN]; float fsum = 0.f;
  #pragma unroll
  for (int w = 0; w < kWIN; ++w) { fw[w] = (ex[w] / den) * conn[h * kWIN + w]; fsum += fw[w]; }
  const float inv = 1.f / (fsum + 1e-9f);

  float orr = 0.f, oii = 0.f;
  #pragma unroll
  for (int w = 0; w < kWIN; ++w) {
    const int l2 = l + w - (kWIN - 1);
    if (l2 >= 0) {
      const size_t rv = (size_t)(b * kL + l2) * kD + h * kHD + 2 * lane;
      const float fwn = fw[w] * inv;
      orr += fwn * v[rv];
      oii += fwn * v[rv + 1];
    }
  }
  ctx[rowq]     = (_Float16)orr;
  ctx[rowq + 1] = (_Float16)oii;
}

extern "C" void kernel_launch(void* const* d_in, const int* in_sizes, int n_in,
                              void* d_out, int out_size, void* d_ws, size_t ws_size,
                              hipStream_t stream) {
  (void)in_sizes; (void)n_in; (void)out_size; (void)ws_size;
  const float* x    = (const float*)d_in[0];
  const float* fcos = (const float*)d_in[1];
  const float* fsin = (const float*)d_in[2];
  const float* Wq   = (const float*)d_in[3];
  const float* Wk   = (const float*)d_in[4];
  const float* Wv   = (const float*)d_in[5];
  const float* Wo   = (const float*)d_in[6];
  const float* W1   = (const float*)d_in[7];
  const float* W2   = (const float*)d_in[8];
  const float* W3   = (const float*)d_in[9];
  float* out = (float*)d_out;

  // Workspace layout (72 MB):
  // [xh: BLD f16][Wqh,Wkh,Wvh,Woh: 4*DD f16][q,k,v: 3*BLD f32][ctx: BLD f16][conn]
  _Float16* xh  = (_Float16*)d_ws;
  _Float16* Wqh = xh  + kBLD;
  _Float16* Wkh = Wqh + kDD;
  _Float16* Wvh = Wkh + kDD;
  _Float16* Woh = Wvh + kDD;
  float*    qb  = (float*)(Woh + kDD);   // 16 MB offset -> aligned
  float*    kb  = qb + kBLD;
  float*    vb  = kb + kBLD;
  _Float16* ctx = (_Float16*)(vb + kBLD);
  float*    conn = (float*)(ctx + kBLD);

  // --- Precision pass: f32 -> f16 for all GEMM operands (single stream pass)
  f32_to_f16_kernel<<<(int)(kBLD / 4 / 256), 256, 0, stream>>>(x,  xh,  (int)(kBLD / 4));
  f32_to_f16_kernel<<<(int)(kDD  / 4 / 256), 256, 0, stream>>>(Wq, Wqh, (int)(kDD / 4));
  f32_to_f16_kernel<<<(int)(kDD  / 4 / 256), 256, 0, stream>>>(Wk, Wkh, (int)(kDD / 4));
  f32_to_f16_kernel<<<(int)(kDD  / 4 / 256), 256, 0, stream>>>(Wv, Wvh, (int)(kDD / 4));
  f32_to_f16_kernel<<<(int)(kDD  / 4 / 256), 256, 0, stream>>>(Wo, Woh, (int)(kDD / 4));

  // --- Fused QKV projection: 2048 waves (32x64 tile each), gridDim.z = Wq/Wk/Wv
  const int waves = (kM / 32) * (kD / 64);        // 2048
  gemm_f16wmma<<<dim3(waves / 4, 1, 3), 128, 0, stream>>>(
      xh, Wqh, Wkh, Wvh, qb, kb, vb, kM, kD, kD);

  gate_kernel<<<1, 256, 0, stream>>>(W1, W2, W3, conn);

  const int attn_threads = kB * kH * kL * 32;     // one wave per (b,h,l)
  attn_kernel<<<attn_threads / 256, 256, 0, stream>>>(
      qb, kb, vb, fcos, fsin, conn, ctx);

  // --- Output projection (ctx already f16)
  gemm_f16wmma<<<dim3(waves / 4, 1, 1), 128, 0, stream>>>(
      ctx, Woh, Woh, Woh, out, out, out, kM, kD, kD);
}
